// Attention_63367947485834
// MI455X (gfx1250) — matrix operand order repaired and checked by reading, not compiled, not run on hardware
//
#include <hip/hip_runtime.h>
#include <hip/hip_bf16.h>

// ---------- types ----------
typedef __attribute__((ext_vector_type(16))) __bf16 v16bf;
typedef __attribute__((ext_vector_type(8)))  __bf16 v8bf;
typedef __attribute__((ext_vector_type(8)))  float  v8f;
typedef __attribute__((ext_vector_type(4)))  unsigned v4u;
typedef __attribute__((ext_vector_type(8)))  unsigned v8u;

union BF16x16 { v16bf v; v8bf h[2]; };

#define WMMA_BF16(a, b, c) \
  __builtin_amdgcn_wmma_f32_16x16x32_bf16(false, (a), false, (b), (short)0, (c), false, false)

static __device__ __forceinline__ __bf16 f2bf(float f) {
  unsigned u = __builtin_bit_cast(unsigned, f);
  unsigned r = u + 0x7FFFu + ((u >> 16) & 1u);   // round-to-nearest-even
  unsigned short hv = (unsigned short)(r >> 16);
  return __builtin_bit_cast(__bf16, hv);
}

// ---------- problem constants ----------
#define BB   2
#define TT   2048
#define DD   2048
#define HH   16
#define DH   128
#define NQKV 6144          // 3*D
#define MR   4096          // B*T
#define TILE_ELEMS (128 * 32)

// =====================================================================
// TDM: issue a 2D tile load (128 rows x 32 bf16 cols, row stride 2048
// elements) from global to LDS via the Tensor Data Mover. Tracked by
// TENSORcnt. 2-operand form => descriptor groups 2/3 are NULL (2D).
// =====================================================================
static __device__ __forceinline__ void tdm_load_tile128x32(
    unsigned lds_byte_addr, const __bf16* gsrc, unsigned tensor_rows) {
  unsigned long ga = (unsigned long)(uintptr_t)gsrc;
  v4u g0;
  g0[0] = 1u;                                  // count = 1 valid descriptor
  g0[1] = lds_byte_addr;                       // lds_addr
  g0[2] = (unsigned)ga;                        // global_addr[31:0]
  g0[3] = (unsigned)(ga >> 32) | (2u << 30);   // global_addr[56:32] | type=2
  v8u g1;
  g1[0] = 1u << 16;                            // data_size = 1 -> 2 bytes
  g1[1] = (DD & 0xFFFFu) << 16;                // tensor_dim0[15:0] @ [31:16]
  g1[2] = (DD >> 16) | ((tensor_rows & 0xFFFFu) << 16);  // dim0 hi | dim1 lo
  g1[3] = (tensor_rows >> 16) | (32u << 16);   // dim1 hi | tile_dim0 = 32
  g1[4] = 128u;                                // tile_dim1 = 128, tile_dim2 = 0
  g1[5] = (unsigned)DD;                        // tensor_dim0_stride lo32
  g1[6] = 0u;                                  // stride hi16, dim1_stride lo16
  g1[7] = 0u;
  asm volatile("tensor_load_to_lds %0, %1" :: "s"(g0), "s"(g1) : "memory");
}

// =====================================================================
// Shared GEMM mainloop: C[128x128] += A[128xK] * W[128xK]^T, bf16 WMMA.
// 4 waves, 64x64 wave tile (16 WMMAs / k-step). TDM-fed double-buffered
// LDS staging with s_wait_tensorcnt pipelining.
// NOTE: double-buffer selection is an *index* offset on a single shared
// base pointer so addrspace inference keeps the reads as ds_load_b128.
// =====================================================================
static __device__ __forceinline__ void gemm_mainloop(
    const __bf16* __restrict__ Amat, const __bf16* __restrict__ Wmat,
    unsigned rowsA, unsigned rowsB, int m0, int n0,
    __bf16* AsBase, __bf16* BsBase,   // shared arrays of 2*TILE_ELEMS each
    int wave, int lane, v8f acc[4][4])
{
  const int g  = lane >> 4;
  const int lr = lane & 15;
  const int wm = (wave & 1) * 64;
  const int wn = (wave >> 1) * 64;

  const unsigned asAddr = (unsigned)(uintptr_t)AsBase;
  const unsigned bsAddr = (unsigned)(uintptr_t)BsBase;

  const __bf16* Abase = Amat + (size_t)m0 * DD;
  const __bf16* Bbase = Wmat + (size_t)n0 * DD;

  const int NK = DD / 32;   // 64 k-steps

  if (threadIdx.x == 0) {   // one wave issues the DMA; EXEC ignored by TDM
    tdm_load_tile128x32(asAddr, Abase, rowsA);
    tdm_load_tile128x32(bsAddr, Bbase, rowsB);
  }

  for (int i = 0; i < NK; ++i) {
    const int boff = (i & 1) * TILE_ELEMS;          // element offset of current buf
    if (threadIdx.x == 0) {
      if (i + 1 < NK) {
        const unsigned nbyte = (unsigned)(((i + 1) & 1) * TILE_ELEMS * 2);
        tdm_load_tile128x32(asAddr + nbyte, Abase + (i + 1) * 32, rowsA);
        tdm_load_tile128x32(bsAddr + nbyte, Bbase + (i + 1) * 32, rowsB);
        __builtin_amdgcn_s_wait_tensorcnt(2);  // current buf pair complete
      } else {
        __builtin_amdgcn_s_wait_tensorcnt(0);  // last tile complete
      }
    }
    __syncthreads();

    BF16x16 a[4], b[4];
#pragma unroll
    for (int mt = 0; mt < 4; ++mt) {
      const __bf16* p = &AsBase[boff + (wm + mt * 16 + lr) * 32];
      a[mt].h[0] = *(const v8bf*)(p + g * 8);        // K = 8g..8g+7
      a[mt].h[1] = *(const v8bf*)(p + 16 + g * 8);   // K = 16+8g..
    }
#pragma unroll
    for (int nt = 0; nt < 4; ++nt) {
      const __bf16* p = &BsBase[boff + (wn + nt * 16 + lr) * 32];
      b[nt].h[0] = *(const v8bf*)(p + g * 8);
      b[nt].h[1] = *(const v8bf*)(p + 16 + g * 8);
    }
#pragma unroll
    for (int mt = 0; mt < 4; ++mt)
#pragma unroll
      for (int nt = 0; nt < 4; ++nt)
        acc[mt][nt] = WMMA_BF16(a[mt].v, b[nt].v, acc[mt][nt]);
    __syncthreads();
  }
}

// =====================================================================
// Kernel 0: f32 -> bf16 cast
// =====================================================================
__global__ void cast_bf16(const float* __restrict__ in, __bf16* __restrict__ out, int n) {
  int i = blockIdx.x * blockDim.x + threadIdx.x;
  if (i < n) out[i] = f2bf(in[i]);
}

// =====================================================================
// Kernel 1: QKV GEMM  qkv[m,n] = sum_k X[m,k]*Wqkv[n,k] + b[n]
// =====================================================================
__global__ __launch_bounds__(128) void qkv_gemm(
    const __bf16* __restrict__ X, const __bf16* __restrict__ W,
    const float* __restrict__ bias,
    __bf16* __restrict__ q_bf, __bf16* __restrict__ k_bf, __bf16* __restrict__ vT_bf,
    float* __restrict__ k_out, float* __restrict__ v_out)
{
  __shared__ __align__(32) __bf16 As[2 * TILE_ELEMS];
  __shared__ __align__(32) __bf16 Bs[2 * TILE_ELEMS];

  const int m0 = blockIdx.y * 128;
  const int n0 = blockIdx.x * 128;
  const int wave = threadIdx.x >> 5;
  const int lane = threadIdx.x & 31;
  const int g  = lane >> 4;
  const int lr = lane & 15;
  const int wm = (wave & 1) * 64;
  const int wn = (wave >> 1) * 64;

  v8f acc[4][4] = {};
  gemm_mainloop(X, W, MR, NQKV, m0, n0, As, Bs, wave, lane, acc);

  // Epilogue: bias + route q/k/v
#pragma unroll
  for (int mt = 0; mt < 4; ++mt) {
#pragma unroll
    for (int nt = 0; nt < 4; ++nt) {
#pragma unroll
      for (int r = 0; r < 8; ++r) {
        int mrow = m0 + wm + mt * 16 + r + 8 * g;   // [0, 4096)
        int ncol = n0 + wn + nt * 16 + lr;          // [0, 6144)
        float val = acc[mt][nt][r] + bias[ncol];
        int bidx  = mrow >> 11;
        int t     = mrow & (TT - 1);
        int hidx  = ncol / 384;
        int inner = ncol - hidx * 384;
        int bh = bidx * HH + hidx;
        if (inner < DH) {
          q_bf[((size_t)bh * TT + t) * DH + inner] = f2bf(val);
        } else if (inner < 2 * DH) {
          size_t o = ((size_t)bh * TT + t) * DH + (inner - DH);
          k_out[o] = val;
          k_bf[o]  = f2bf(val);
        } else {
          int c = inner - 2 * DH;
          v_out[((size_t)bh * TT + t) * DH + c] = val;
          vT_bf[((size_t)bh * DH + c) * TT + t] = f2bf(val);   // transposed for P@V
        }
      }
    }
  }
}

// =====================================================================
// Kernel 2: causal flash attention, one wave per (b,h, 32-row q tile).
// =====================================================================
__global__ __launch_bounds__(32) void attn_kernel(
    const __bf16* __restrict__ q_bf, const __bf16* __restrict__ k_bf,
    const __bf16* __restrict__ vT_bf, __bf16* __restrict__ attn_bf)
{
  __shared__ __align__(32) __bf16 Pl[32 * 40];   // 32x32 P tile, padded stride

  const int bid = blockIdx.x;
  const int bh  = bid >> 6;
  const int qt  = bid & 63;
  const int q0  = qt * 32;
  const int bidx = bh >> 4;
  const int hidx = bh & 15;
  const int lane = threadIdx.x;
  const int g  = lane >> 4;
  const int lr = lane & 15;
  const float sc = 0.08838834764831845f;  // 1/sqrt(128)

  const __bf16* Qb = q_bf  + (size_t)bh * TT * DH;
  const __bf16* Kb = k_bf  + (size_t)bh * TT * DH;
  const __bf16* Vb = vT_bf + (size_t)bh * DH * TT;

  BF16x16 aq[2][4];
#pragma unroll
  for (int mt = 0; mt < 2; ++mt)
#pragma unroll
    for (int ks = 0; ks < 4; ++ks) {
      const __bf16* p = Qb + (size_t)(q0 + mt * 16 + lr) * DH + ks * 32;
      aq[mt][ks].h[0] = *(const v8bf*)(p + g * 8);
      aq[mt][ks].h[1] = *(const v8bf*)(p + 16 + g * 8);
    }

  v8f o[2][8] = {};
  float rmax[2][8], rsum[2][8];
#pragma unroll
  for (int mt = 0; mt < 2; ++mt)
#pragma unroll
    for (int r = 0; r < 8; ++r) { rmax[mt][r] = -1e30f; rsum[mt][r] = 0.f; }

  for (int kt = 0; kt <= qt; ++kt) {
    const int kv0 = kt * 32;

    v8f s[2][2] = {};
#pragma unroll
    for (int ks = 0; ks < 4; ++ks) {
      BF16x16 bk[2];
#pragma unroll
      for (int nt = 0; nt < 2; ++nt) {
        const __bf16* p = Kb + (size_t)(kv0 + nt * 16 + lr) * DH + ks * 32;
        bk[nt].h[0] = *(const v8bf*)(p + g * 8);
        bk[nt].h[1] = *(const v8bf*)(p + 16 + g * 8);
      }
#pragma unroll
      for (int mt = 0; mt < 2; ++mt)
#pragma unroll
        for (int nt = 0; nt < 2; ++nt)
          s[mt][nt] = WMMA_BF16(aq[mt][ks].v, bk[nt].v, s[mt][nt]);
    }

#pragma unroll
    for (int mt = 0; mt < 2; ++mt) {
#pragma unroll
      for (int r = 0; r < 8; ++r) {
        const int qg = q0 + mt * 16 + r + 8 * g;
        float s0 = s[mt][0][r], s1 = s[mt][1][r];
        const int kg0 = kv0 + lr, kg1 = kv0 + 16 + lr;
        s0 = (kg0 <= qg) ? s0 * sc : -__builtin_inff();
        s1 = (kg1 <= qg) ? s1 * sc : -__builtin_inff();
        float mx = fmaxf(s0, s1);
#pragma unroll
        for (int off = 1; off < 16; off <<= 1) mx = fmaxf(mx, __shfl_xor(mx, off, 32));
        const float mnew = fmaxf(rmax[mt][r], mx);
        const float corr = __expf(rmax[mt][r] - mnew);
        const float p0 = __expf(s0 - mnew);
        const float p1 = __expf(s1 - mnew);
        float ps = p0 + p1;
#pragma unroll
        for (int off = 1; off < 16; off <<= 1) ps += __shfl_xor(ps, off, 32);
        rsum[mt][r] = rsum[mt][r] * corr + ps;
        rmax[mt][r] = mnew;
#pragma unroll
        for (int vn = 0; vn < 8; ++vn) o[mt][vn][r] *= corr;
        const int m = mt * 16 + r + 8 * g;
        Pl[m * 40 + lr]      = f2bf(p0);
        Pl[m * 40 + 16 + lr] = f2bf(p1);
      }
    }
    __syncthreads();

    BF16x16 ap[2];
#pragma unroll
    for (int mt = 0; mt < 2; ++mt) {
      const __bf16* p = &Pl[(mt * 16 + lr) * 40];
      ap[mt].h[0] = *(const v8bf*)(p + g * 8);
      ap[mt].h[1] = *(const v8bf*)(p + 16 + g * 8);
    }

#pragma unroll
    for (int vn = 0; vn < 8; ++vn) {
      BF16x16 bv;
      const __bf16* p = Vb + (size_t)(vn * 16 + lr) * TT + kv0;
      bv.h[0] = *(const v8bf*)(p + g * 8);
      bv.h[1] = *(const v8bf*)(p + 16 + g * 8);
#pragma unroll
      for (int mt = 0; mt < 2; ++mt)
        o[mt][vn] = WMMA_BF16(ap[mt].v, bv.v, o[mt][vn]);
    }
    __syncthreads();
  }

#pragma unroll
  for (int mt = 0; mt < 2; ++mt) {
#pragma unroll
    for (int r = 0; r < 8; ++r) {
      const float inv = 1.0f / rsum[mt][r];
      const int t = q0 + mt * 16 + r + 8 * g;
      const size_t base = ((size_t)bidx * TT + t) * DD + hidx * DH;
#pragma unroll
      for (int vn = 0; vn < 8; ++vn)
        attn_bf[base + vn * 16 + lr] = f2bf(o[mt][vn][r] * inv);
    }
  }
}

// =====================================================================
// Kernel 3: output projection
// =====================================================================
__global__ __launch_bounds__(128) void proj_gemm(
    const __bf16* __restrict__ A, const __bf16* __restrict__ W,
    const float* __restrict__ bias, float* __restrict__ out)
{
  __shared__ __align__(32) __bf16 As[2 * TILE_ELEMS];
  __shared__ __align__(32) __bf16 Bs[2 * TILE_ELEMS];

  const int m0 = blockIdx.y * 128;
  const int n0 = blockIdx.x * 128;
  const int wave = threadIdx.x >> 5;
  const int lane = threadIdx.x & 31;
  const int g  = lane >> 4;
  const int lr = lane & 15;
  const int wm = (wave & 1) * 64;
  const int wn = (wave >> 1) * 64;

  v8f acc[4][4] = {};
  gemm_mainloop(A, W, MR, DD, m0, n0, As, Bs, wave, lane, acc);

#pragma unroll
  for (int mt = 0; mt < 4; ++mt)
#pragma unroll
    for (int nt = 0; nt < 4; ++nt)
#pragma unroll
      for (int r = 0; r < 8; ++r) {
        int mrow = m0 + wm + mt * 16 + r + 8 * g;
        int ncol = n0 + wn + nt * 16 + lr;
        out[(size_t)mrow * DD + ncol] = acc[mt][nt][r] + bias[ncol];
      }
}

// =====================================================================
// launch
// =====================================================================
extern "C" void kernel_launch(void* const* d_in, const int* in_sizes, int n_in,
                              void* d_out, int out_size, void* d_ws, size_t ws_size,
                              hipStream_t stream) {
  (void)in_sizes; (void)n_in; (void)out_size; (void)ws_size;
  const float* x     = (const float*)d_in[0];
  const float* wqkv  = (const float*)d_in[1];
  const float* bqkv  = (const float*)d_in[2];
  const float* wproj = (const float*)d_in[3];
  const float* bproj = (const float*)d_in[4];

  float* out_f = (float*)d_out;
  float* k_out = out_f + (size_t)BB * TT * DD;
  float* v_out = k_out + (size_t)BB * TT * DD;

  const size_t nX  = (size_t)BB * TT * DD;
  const size_t nWq = (size_t)NQKV * DD;
  const size_t nWp = (size_t)DD * DD;

  char* ws = (char*)d_ws;
  size_t off = 0;
  auto alloc = [&](size_t bytes) -> void* {
    void* p = ws + off;
    off += (bytes + 255) & ~(size_t)255;
    return p;
  };
  __bf16* x_bf  = (__bf16*)alloc(nX * 2);
  __bf16* wq_bf = (__bf16*)alloc(nWq * 2);
  __bf16* wp_bf = (__bf16*)alloc(nWp * 2);
  __bf16* q_bf  = (__bf16*)alloc(nX * 2);
  __bf16* k_bf  = (__bf16*)alloc(nX * 2);
  __bf16* vT_bf = (__bf16*)alloc(nX * 2);
  __bf16* at_bf = (__bf16*)alloc(nX * 2);

  cast_bf16<<<(int)((nX  + 255) / 256), 256, 0, stream>>>(x,     x_bf,  (int)nX);
  cast_bf16<<<(int)((nWq + 255) / 256), 256, 0, stream>>>(wqkv,  wq_bf, (int)nWq);
  cast_bf16<<<(int)((nWp + 255) / 256), 256, 0, stream>>>(wproj, wp_bf, (int)nWp);

  qkv_gemm<<<dim3(NQKV / 128, MR / 128), 128, 0, stream>>>(
      x_bf, wq_bf, bqkv, q_bf, k_bf, vT_bf, k_out, v_out);

  attn_kernel<<<BB * HH * (TT / 32), 32, 0, stream>>>(q_bf, k_bf, vT_bf, at_bf);

  proj_gemm<<<dim3(DD / 128, MR / 128), 128, 0, stream>>>(at_bf, wp_bf, bproj, out_f);
}
